// LSTMCell_3358664426297
// MI455X (gfx1250) — matrix operand order repaired and checked by reading, not compiled, not run on hardware
//
#include <hip/hip_runtime.h>
#include <hip/hip_bf16.h>

typedef __attribute__((ext_vector_type(16))) __bf16          v16bf;
typedef __attribute__((ext_vector_type(8)))  float           v8f;
typedef __attribute__((ext_vector_type(16))) unsigned short  v16us;

// Pack two fp32 -> packed bf16 dword (truncation) with a single v_perm_b32.
__device__ __forceinline__ unsigned pk_bf16(float lo, float hi) {
    union { float f; unsigned u; } a, b;
    a.f = lo; b.f = hi;
    // selector bytes pick {hi[31:24],hi[23:16],lo[31:24],lo[23:16]}
    return __builtin_amdgcn_perm(b.u, a.u, 0x07060302u);
}
__device__ __forceinline__ float fsigmoid(float x) {
    return 1.0f / (1.0f + __expf(-x));
}
__device__ __forceinline__ float ftanh(float x) {
    float e = __expf(-2.0f * x);
    return (1.0f - e) / (1.0f + e);
}
__device__ __forceinline__ void wait_async0() {
#if __has_builtin(__builtin_amdgcn_s_wait_asynccnt)
    __builtin_amdgcn_s_wait_asynccnt(0);
#else
    asm volatile("s_wait_asynccnt 0x0" ::: "memory");
#endif
}

// ---------------------------------------------------------------------------
// Pass 1: pack fp32 [rows][512]|[rows][512] -> bf16 [rows][1024] (K-concat).
// One dword (2 bf16) per thread-iteration, one v_perm each.
// ---------------------------------------------------------------------------
__global__ __launch_bounds__(256)
void cvt_pack(const float* __restrict__ s0, const float* __restrict__ s1,
              unsigned* __restrict__ dst, int rows)
{
    const int nd = rows * 512;                      // dwords of output
    for (int p = blockIdx.x * 256 + threadIdx.x; p < nd; p += gridDim.x * 256) {
        const int row = p >> 9;
        const int cp  = p & 511;                    // dword col in [0,512)
        const float* s = (cp < 256) ? (s0 + (size_t)row * 512 + cp * 2)
                                    : (s1 + (size_t)row * 512 + (cp - 256) * 2);
        float2 v = *(const float2*)s;
        dst[p] = pk_bf16(v.x, v.y);
    }
}

// ---------------------------------------------------------------------------
// Main GEMM+LSTM kernel, pre-packed bf16 operands.
// Ab: bf16 [16384][1024] (x|h), Wb: bf16 [2048][1024] (Wi|Wh).
// Grid (128 M-tiles x 16 N-tiles), 256 threads = 8 waves; wave owns a 16-row
// strip and 8 16x16 accum tiles (4 gates x 2 column-halves).
// B tile staged in LDS via global_load_async_to_lds_b128, double buffered.
// ---------------------------------------------------------------------------
__global__ __launch_bounds__(256)
void lstm_cell_wmma_pre(const unsigned short* __restrict__ Ab,
                        const unsigned short* __restrict__ Wb,
                        const float* __restrict__ c,
                        const float* __restrict__ bi,
                        const float* __restrict__ bh,
                        float* __restrict__ out)
{
    constexpr int Hd = 512;
    constexpr int Kd = 1024;
    constexpr int Bd = 16384;

    __shared__ unsigned short Bs[2][128][32];       // 16 KB, double buffered

    const int tid    = threadIdx.x;
    const int wave   = tid >> 5;
    const int lane   = tid & 31;
    const int lane15 = lane & 15;
    const int laneHi = lane >> 4;
    const int hc0    = blockIdx.y * 32;
    const int row0   = blockIdx.x * 128 + wave * 16;

    v8f acc[8];
#pragma unroll
    for (int s = 0; s < 8; ++s) acc[s] = (v8f){0,0,0,0,0,0,0,0};

    // staging: 2 threads per tile row, 32 bytes each (2x async b128)
    const int srow  = tid >> 1;                                  // 0..127
    const int wrow  = (srow >> 5) * Hd + hc0 + (srow & 31);      // weight row
    const int sKoff = (tid & 1) * 16;                            // bf16 elems
    const unsigned short* gRow = Wb + (size_t)wrow * Kd + sKoff;
    unsigned ldsOff[2];
    ldsOff[0] = (unsigned)(size_t)&Bs[0][srow][sKoff];
    ldsOff[1] = (unsigned)(size_t)&Bs[1][srow][sKoff];

    // A fragment base: lane -> K{b..b+15}, b = laneHi*16
    const unsigned short* aRow = Ab + (size_t)(row0 + lane15) * Kd + laneHi * 16;

    auto stage = [&](int ks, int buf) {
        unsigned long long ga = (unsigned long long)(size_t)(gRow + ks);
        asm volatile(
            "global_load_async_to_lds_b128 %0, %1, off\n\t"
            "global_load_async_to_lds_b128 %0, %1, off offset:16"
            :: "v"(ldsOff[buf]), "v"(ga) : "memory");
    };

    stage(0, 0);
    for (int ks = 0; ks < Kd; ks += 32) {
        const int buf = (ks >> 5) & 1;
        wait_async0();
        __syncthreads();
        if (ks + 32 < Kd) stage(ks + 32, buf ^ 1);   // overlap next copy

        // A fragment: two b128 loads straight from bf16 (no conversion)
        union { uint4 q[2]; v16us v; } aa;
        aa.q[0] = *(const uint4*)(aRow + ks);
        aa.q[1] = *(const uint4*)(aRow + ks + 8);
        v16bf af = __builtin_bit_cast(v16bf, aa.v);

        // 8 WMMA tiles, B-fragment ds_loads software-pipelined
        v16us bcur = *(const v16us*)&Bs[buf][lane15][laneHi * 16];
#pragma unroll
        for (int s = 0; s < 8; ++s) {
            v16us bnxt;
            if (s < 7) bnxt = *(const v16us*)&Bs[buf][(s + 1) * 16 + lane15][laneHi * 16];
            v16bf bf = __builtin_bit_cast(v16bf, bcur);
            acc[s] = __builtin_amdgcn_wmma_f32_16x16x32_bf16(
                false, af, false, bf, (short)0, acc[s], false, false);
            if (s < 7) bcur = bnxt;
        }
    }

    // fused epilogue
#pragma unroll
    for (int halfn = 0; halfn < 2; ++halfn) {
        const int col = hc0 + halfn * 16 + lane15;
        const float b_i = bi[        col] + bh[        col];
        const float b_f = bi[ 512 + col] + bh[ 512 + col];
        const float b_g = bi[1024 + col] + bh[1024 + col];
        const float b_o = bi[1536 + col] + bh[1536 + col];
        v8f iv = acc[0 + halfn], fv = acc[2 + halfn];
        v8f gv = acc[4 + halfn], ov = acc[6 + halfn];
#pragma unroll
        for (int v = 0; v < 8; ++v) {
            const int row = row0 + v + laneHi * 8;
            const size_t idx = (size_t)row * Hd + col;
            float ig = fsigmoid(iv[v] + b_i);
            float fg = fsigmoid(fv[v] + b_f);
            float gg = ftanh   (gv[v] + b_g);
            float og = fsigmoid(ov[v] + b_o);
            float nc = fg * c[idx] + ig * gg;
            float nh = og * ftanh(nc);
            out[idx]                   = nh;
            out[(size_t)Bd * Hd + idx] = nc;
        }
    }
}

// ---------------------------------------------------------------------------
// Fallback (ws too small): fused in-kernel conversion, perm-based packing.
// ---------------------------------------------------------------------------
__global__ __launch_bounds__(256)
void lstm_cell_wmma_fb(const float* __restrict__ x,  const float* __restrict__ h,
                       const float* __restrict__ c,  const float* __restrict__ Wi,
                       const float* __restrict__ bi, const float* __restrict__ Wh,
                       const float* __restrict__ bh, float* __restrict__ out)
{
    constexpr int Hd = 512;
    constexpr int Bd = 16384;
    __shared__ unsigned short Bs[2][128][32];

    const int tid    = threadIdx.x;
    const int wave   = tid >> 5;
    const int lane   = tid & 31;
    const int lane15 = lane & 15;
    const int laneHi = lane >> 4;
    const int hc0    = blockIdx.y * 32;
    const int row0   = blockIdx.x * 128 + wave * 16;

    v8f acc[8];
#pragma unroll
    for (int s = 0; s < 8; ++s) acc[s] = (v8f){0,0,0,0,0,0,0,0};

    const int srow  = tid >> 1;
    const int wrow  = (srow >> 5) * Hd + hc0 + (srow & 31);
    const int khalf = (tid & 1) * 16;

    const float* aRowX = x + (size_t)(row0 + lane15) * Hd;
    const float* aRowH = h + (size_t)(row0 + lane15) * Hd;
    const int    aOff  = laneHi * 16;

    for (int ks = 0; ks < 1024; ks += 32) {
        const int buf = (ks >> 5) & 1;
        const int kk  = ks & 511;
        {
            const float* wsrc = ((ks < 512) ? Wi : Wh) + (size_t)wrow * Hd + kk + khalf;
            float4 w0 = ((const float4*)wsrc)[0];
            float4 w1 = ((const float4*)wsrc)[1];
            float4 w2 = ((const float4*)wsrc)[2];
            float4 w3 = ((const float4*)wsrc)[3];
            uint4 q0, q1;
            q0.x = pk_bf16(w0.x, w0.y); q0.y = pk_bf16(w0.z, w0.w);
            q0.z = pk_bf16(w1.x, w1.y); q0.w = pk_bf16(w1.z, w1.w);
            q1.x = pk_bf16(w2.x, w2.y); q1.y = pk_bf16(w2.z, w2.w);
            q1.z = pk_bf16(w3.x, w3.y); q1.w = pk_bf16(w3.z, w3.w);
            uint4* dst = (uint4*)&Bs[buf][srow][khalf];
            dst[0] = q0; dst[1] = q1;
        }
        __syncthreads();

        const float* asrc = ((ks < 512) ? aRowX : aRowH) + kk + aOff;
        float4 a0 = ((const float4*)asrc)[0];
        float4 a1 = ((const float4*)asrc)[1];
        float4 a2 = ((const float4*)(asrc + 8))[0];
        float4 a3 = ((const float4*)(asrc + 8))[1];
        union { unsigned d[8]; v16us v; } aa;
        aa.d[0] = pk_bf16(a0.x, a0.y); aa.d[1] = pk_bf16(a0.z, a0.w);
        aa.d[2] = pk_bf16(a1.x, a1.y); aa.d[3] = pk_bf16(a1.z, a1.w);
        aa.d[4] = pk_bf16(a2.x, a2.y); aa.d[5] = pk_bf16(a2.z, a2.w);
        aa.d[6] = pk_bf16(a3.x, a3.y); aa.d[7] = pk_bf16(a3.z, a3.w);
        v16bf af = __builtin_bit_cast(v16bf, aa.v);

        v16us bcur = *(const v16us*)&Bs[buf][lane15][laneHi * 16];
#pragma unroll
        for (int s = 0; s < 8; ++s) {
            v16us bnxt;
            if (s < 7) bnxt = *(const v16us*)&Bs[buf][(s + 1) * 16 + lane15][laneHi * 16];
            v16bf bf = __builtin_bit_cast(v16bf, bcur);
            acc[s] = __builtin_amdgcn_wmma_f32_16x16x32_bf16(
                false, af, false, bf, (short)0, acc[s], false, false);
            if (s < 7) bcur = bnxt;
        }
    }

#pragma unroll
    for (int halfn = 0; halfn < 2; ++halfn) {
        const int col = hc0 + halfn * 16 + lane15;
        const float b_i = bi[        col] + bh[        col];
        const float b_f = bi[ 512 + col] + bh[ 512 + col];
        const float b_g = bi[1024 + col] + bh[1024 + col];
        const float b_o = bi[1536 + col] + bh[1536 + col];
        v8f iv = acc[0 + halfn], fv = acc[2 + halfn];
        v8f gv = acc[4 + halfn], ov = acc[6 + halfn];
#pragma unroll
        for (int v = 0; v < 8; ++v) {
            const int row = row0 + v + laneHi * 8;
            const size_t idx = (size_t)row * Hd + col;
            float ig = fsigmoid(iv[v] + b_i);
            float fg = fsigmoid(fv[v] + b_f);
            float gg = ftanh   (gv[v] + b_g);
            float og = fsigmoid(ov[v] + b_o);
            float nc = fg * c[idx] + ig * gg;
            float nh = og * ftanh(nc);
            out[idx]                   = nh;
            out[(size_t)Bd * Hd + idx] = nc;
        }
    }
}

extern "C" void kernel_launch(void* const* d_in, const int* in_sizes, int n_in,
                              void* d_out, int out_size, void* d_ws, size_t ws_size,
                              hipStream_t stream) {
    const float* x  = (const float*)d_in[0];
    const float* h  = (const float*)d_in[1];
    const float* c  = (const float*)d_in[2];
    const float* Wi = (const float*)d_in[3];
    const float* bi = (const float*)d_in[4];
    const float* Wh = (const float*)d_in[5];
    const float* bh = (const float*)d_in[6];
    float* out = (float*)d_out;

    const size_t nA = (size_t)16384 * 1024;      // bf16 elements
    const size_t nW = (size_t)2048 * 1024;
    const size_t need = (nA + nW) * sizeof(unsigned short);   // 36 MB

    dim3 grid(16384 / 128, 512 / 32), block(256);

    if (ws_size >= need) {
        unsigned short* Ab = (unsigned short*)d_ws;
        unsigned short* Wb = Ab + nA;
        cvt_pack<<<dim3((unsigned)((nA / 2 + 255) / 256)), block, 0, stream>>>(
            x, h, (unsigned*)Ab, 16384);
        cvt_pack<<<dim3((unsigned)((nW / 2 + 255) / 256)), block, 0, stream>>>(
            Wi, Wh, (unsigned*)Wb, 2048);
        lstm_cell_wmma_pre<<<grid, block, 0, stream>>>(Ab, Wb, c, bi, bh, out);
    } else {
        lstm_cell_wmma_fb<<<grid, block, 0, stream>>>(x, h, c, Wi, bi, Wh, bh, out);
    }
}
